// LSTMModel_57964878626944
// MI455X (gfx1250) — compile-verified, matrix operand order checked by reading, not run on hardware
//
#include <hip/hip_runtime.h>
#include <hip/hip_bf16.h>

// ---------------------------------------------------------------------------
// Fused 2-layer LSTM (B=4096, S=512, F=1, H=64) + MLP head (64->32->24).
// One workgroup = 16 batch rows (WMMA M=16), 4 waves; wave w owns gate block w
// (i/f/g/o) of BOTH layers. h1/h2 streamed per-timestep through LDS in WMMA
// A-fragment layout; c state in registers. v_wmma_f32_16x16x32_f16 everywhere.
// Elementwise cell uses hardware v_exp_f32 / v_rcp_f32 (no precise division).
// ---------------------------------------------------------------------------

typedef __attribute__((ext_vector_type(16))) _Float16 v16h;
typedef __attribute__((ext_vector_type(8)))  _Float16 v8h;
typedef __attribute__((ext_vector_type(8)))  float    v8f;

#define DEVI __device__ __forceinline__

// sigmoid(x) = rcp(1 + exp(-x))        : v_exp + v_add + v_rcp
DEVI float sig_(float x) {
    return __builtin_amdgcn_rcpf(1.0f + __expf(-x));
}
// tanh(x) = 1 - 2*rcp(exp(2x) + 1)     : v_exp + v_fma + v_rcp (+clamp)
DEVI float tanh_(float x) {
    x = fminf(fmaxf(x, -15.0f), 15.0f);
    float e = __expf(2.0f * x);
    return 1.0f - 2.0f * __builtin_amdgcn_rcpf(e + 1.0f);
}

// A/B fragment element mapping for 16-bit 16x32 (A) / 32x16 (B) tiles:
//   lane = rc + 16*k'[3]   (rc = row for A, col for B; 0..15)
//   elem i = (k'[4]<<3) | k'[2:0]
// inverse: k' = (i[3]<<4) | (lane[4]<<3) | i[2:0]
DEVI int kprime_of(int lane, int i) {
    return ((i >> 3) << 4) | (((lane >> 4) & 1) << 3) | (i & 7);
}

__global__ __launch_bounds__(128)
void lstm2_fused_kernel(const float* __restrict__ x,
                        const float* __restrict__ Wih1, const float* __restrict__ Whh1,
                        const float* __restrict__ bih1, const float* __restrict__ bhh1,
                        const float* __restrict__ Wih2, const float* __restrict__ Whh2,
                        const float* __restrict__ bih2, const float* __restrict__ bhh2,
                        const float* __restrict__ Wd1,  const float* __restrict__ bd1,
                        const float* __restrict__ Wd2,  const float* __restrict__ bd2,
                        float* __restrict__ out, int S)
{
    // LDS: 32KB W_ih2 B-frags + 16KB z + 4KB packed h (4 K-chunks) + 4KB x + 1KB y
    __shared__ _Float16 sB2[2 * 16 * 512];   // W_ih2, B layout, [kc][ntile][lane*16+i]
    __shared__ float    sZ[16 * 256];        // gate pre-activations [row][gate]
    __shared__ _Float16 sHA[4 * 512];        // [h1 | h2] packed A layout, K-chunks 0..3
    __shared__ float    sX[64 * 16];         // x chunk [tmod][row]
    __shared__ _Float16 sY[512];             // head intermediate, A layout (K=32)

    const int tid   = threadIdx.x;
    const int w     = tid >> 5;          // wave id 0..3 == gate block (i,f,g,o)
    const int lane  = tid & 31;
    const int n     = lane & 15;         // col within 16-wide N tile
    const int hi    = lane >> 4;
    const int row0  = hi * 8;            // C/D: vgpr r holds row r+row0
    const int bbase = blockIdx.x * 16;

    // ---- pack W_ih2 into LDS as B fragments (layer-2 K-chunks 0,1) ----
    for (int s = tid; s < 2 * 16 * 512; s += 128) {
        int i   = s & 15;
        int L   = (s >> 4) & 31;
        int ntg = (s >> 9) & 15;
        int kc  = s >> 13;
        int kp  = kprime_of(L, i);
        int g   = ntg * 16 + (L & 15);
        sB2[s] = (_Float16)Wih2[g * 64 + kc * 32 + kp];
    }
    for (int s = tid; s < 4 * 512; s += 128) sHA[s] = (_Float16)0.0f;

    // ---- per-wave register B fragments: W_hh1, W_hh2 ----
    v16h whh1[4][2], whh2[4][2];
#pragma unroll
    for (int nt = 0; nt < 4; ++nt) {
        const int gate = w * 64 + nt * 16 + n;
#pragma unroll
        for (int kc = 0; kc < 2; ++kc) {
            v16h b1, b2;
#pragma unroll
            for (int i = 0; i < 16; ++i) {
                int kp = kprime_of(lane, i);
                b1[i] = (_Float16)Whh1[gate * 64 + kc * 32 + kp];
                b2[i] = (_Float16)Whh2[gate * 64 + kc * 32 + kp];
            }
            whh1[nt][kc] = b1;
            whh2[nt][kc] = b2;
        }
    }
    // per-lane gate scalars (F=1 input projection folded into C init)
    float wf1[4], bf1[4], bf2[4];
#pragma unroll
    for (int nt = 0; nt < 4; ++nt) {
        const int gate = w * 64 + nt * 16 + n;
        wf1[nt] = Wih1[gate];
        bf1[nt] = bih1[gate] + bhh1[gate];
        bf2[nt] = bih2[gate] + bhh2[gate];
    }

    // ---- elementwise ownership: thread -> (batch row eb, hidden ej0..ej0+7) ----
    const int eb  = tid >> 3;
    const int ej0 = (tid & 7) * 8;
    float c1[8], c2[8];
#pragma unroll
    for (int q = 0; q < 8; ++q) { c1[q] = 0.0f; c2[q] = 0.0f; }

    __syncthreads();

    for (int t = 0; t < S; ++t) {
        if ((t & 63) == 0) {   // stage next 64 timesteps of x into LDS
            __syncthreads();
            const int r  = tid >> 3;
            const int tq = (tid & 7) * 8;
            const float* xp = &x[(size_t)(bbase + r) * S + t + tq];
#pragma unroll
            for (int u = 0; u < 8; ++u) sX[(tq + u) * 16 + r] = xp[u];
            __syncthreads();
        }
        const int tm = t & 63;

        float xr[8];
        {
            const float4 xa = *(const float4*)&sX[tm * 16 + row0];
            const float4 xb = *(const float4*)&sX[tm * 16 + row0 + 4];
            xr[0] = xa.x; xr[1] = xa.y; xr[2] = xa.z; xr[3] = xa.w;
            xr[4] = xb.x; xr[5] = xb.y; xr[6] = xb.z; xr[7] = xb.w;
        }

        // ---------------- layer 1: z1 = xp1 + h1 @ Whh1^T ----------------
        v16h a0 = *(const v16h*)&sHA[0 * 512 + lane * 16];
        v16h a1 = *(const v16h*)&sHA[1 * 512 + lane * 16];
#pragma unroll
        for (int nt = 0; nt < 4; ++nt) {
            v8f c;
#pragma unroll
            for (int r = 0; r < 8; ++r) c[r] = xr[r] * wf1[nt] + bf1[nt];
            c = __builtin_amdgcn_wmma_f32_16x16x32_f16(false, a0, false, whh1[nt][0],
                                                       (short)0, c, false, false);
            c = __builtin_amdgcn_wmma_f32_16x16x32_f16(false, a1, false, whh1[nt][1],
                                                       (short)0, c, false, false);
            const int g0 = w * 64 + nt * 16 + n;
#pragma unroll
            for (int r = 0; r < 8; ++r) sZ[(r + row0) * 256 + g0] = c[r];
        }
        __syncthreads();

        // ---------------- elementwise LSTM cell, layer 1 ----------------
        {
            const float4 iA = *(const float4*)&sZ[eb * 256 +   0 + ej0];
            const float4 iB = *(const float4*)&sZ[eb * 256 +   0 + ej0 + 4];
            const float4 fA = *(const float4*)&sZ[eb * 256 +  64 + ej0];
            const float4 fB = *(const float4*)&sZ[eb * 256 +  64 + ej0 + 4];
            const float4 gA = *(const float4*)&sZ[eb * 256 + 128 + ej0];
            const float4 gB = *(const float4*)&sZ[eb * 256 + 128 + ej0 + 4];
            const float4 oA = *(const float4*)&sZ[eb * 256 + 192 + ej0];
            const float4 oB = *(const float4*)&sZ[eb * 256 + 192 + ej0 + 4];
            const float iv[8] = {iA.x, iA.y, iA.z, iA.w, iB.x, iB.y, iB.z, iB.w};
            const float fv[8] = {fA.x, fA.y, fA.z, fA.w, fB.x, fB.y, fB.z, fB.w};
            const float gv[8] = {gA.x, gA.y, gA.z, gA.w, gB.x, gB.y, gB.z, gB.w};
            const float ov[8] = {oA.x, oA.y, oA.z, oA.w, oB.x, oB.y, oB.z, oB.w};
            v8h hv;
#pragma unroll
            for (int q = 0; q < 8; ++q) {
                c1[q] = sig_(fv[q]) * c1[q] + sig_(iv[q]) * tanh_(gv[q]);
                hv[q] = (_Float16)(sig_(ov[q]) * tanh_(c1[q]));
            }
            const int chunk = ej0 >> 5, kp = ej0 & 31;
            const int L  = eb + (((kp >> 3) & 1) << 4);
            const int ii = ((kp >> 4) << 3);
            *(v8h*)&sHA[chunk * 512 + L * 16 + ii] = hv;   // repack h1_t (A layout)
        }
        __syncthreads();

        // ------- layer 2: z2 = [h1_t|h2_{t-1}] @ [Wih2;Whh2]^T + bias -------
        a0 = *(const v16h*)&sHA[0 * 512 + lane * 16];
        a1 = *(const v16h*)&sHA[1 * 512 + lane * 16];
        v16h a2 = *(const v16h*)&sHA[2 * 512 + lane * 16];
        v16h a3 = *(const v16h*)&sHA[3 * 512 + lane * 16];
#pragma unroll
        for (int nt = 0; nt < 4; ++nt) {
            const int ntg = w * 4 + nt;
            v8f c;
#pragma unroll
            for (int r = 0; r < 8; ++r) c[r] = bf2[nt];
            v16h b0 = *(const v16h*)&sB2[(0 * 16 + ntg) * 512 + lane * 16];
            v16h b1 = *(const v16h*)&sB2[(1 * 16 + ntg) * 512 + lane * 16];
            c = __builtin_amdgcn_wmma_f32_16x16x32_f16(false, a0, false, b0,
                                                       (short)0, c, false, false);
            c = __builtin_amdgcn_wmma_f32_16x16x32_f16(false, a1, false, b1,
                                                       (short)0, c, false, false);
            c = __builtin_amdgcn_wmma_f32_16x16x32_f16(false, a2, false, whh2[nt][0],
                                                       (short)0, c, false, false);
            c = __builtin_amdgcn_wmma_f32_16x16x32_f16(false, a3, false, whh2[nt][1],
                                                       (short)0, c, false, false);
            const int g0 = w * 64 + nt * 16 + n;
#pragma unroll
            for (int r = 0; r < 8; ++r) sZ[(r + row0) * 256 + g0] = c[r];
        }
        __syncthreads();

        // ---------------- elementwise LSTM cell, layer 2 ----------------
        {
            const float4 iA = *(const float4*)&sZ[eb * 256 +   0 + ej0];
            const float4 iB = *(const float4*)&sZ[eb * 256 +   0 + ej0 + 4];
            const float4 fA = *(const float4*)&sZ[eb * 256 +  64 + ej0];
            const float4 fB = *(const float4*)&sZ[eb * 256 +  64 + ej0 + 4];
            const float4 gA = *(const float4*)&sZ[eb * 256 + 128 + ej0];
            const float4 gB = *(const float4*)&sZ[eb * 256 + 128 + ej0 + 4];
            const float4 oA = *(const float4*)&sZ[eb * 256 + 192 + ej0];
            const float4 oB = *(const float4*)&sZ[eb * 256 + 192 + ej0 + 4];
            const float iv[8] = {iA.x, iA.y, iA.z, iA.w, iB.x, iB.y, iB.z, iB.w};
            const float fv[8] = {fA.x, fA.y, fA.z, fA.w, fB.x, fB.y, fB.z, fB.w};
            const float gv[8] = {gA.x, gA.y, gA.z, gA.w, gB.x, gB.y, gB.z, gB.w};
            const float ov[8] = {oA.x, oA.y, oA.z, oA.w, oB.x, oB.y, oB.z, oB.w};
            v8h hv;
#pragma unroll
            for (int q = 0; q < 8; ++q) {
                c2[q] = sig_(fv[q]) * c2[q] + sig_(iv[q]) * tanh_(gv[q]);
                hv[q] = (_Float16)(sig_(ov[q]) * tanh_(c2[q]));
            }
            const int chunk = 2 + (ej0 >> 5), kp = ej0 & 31;
            const int L  = eb + (((kp >> 3) & 1) << 4);
            const int ii = ((kp >> 4) << 3);
            *(v8h*)&sHA[chunk * 512 + L * 16 + ii] = hv;   // repack h2_t
        }
        __syncthreads();
    }

    // ---------------- head: y = relu(h2 @ Wd1^T + b); out = y @ Wd2^T + b ----
    if (w == 0) {
        v16h a2 = *(const v16h*)&sHA[2 * 512 + lane * 16];
        v16h a3 = *(const v16h*)&sHA[3 * 512 + lane * 16];
#pragma unroll
        for (int nt = 0; nt < 2; ++nt) {
            const int gcol = nt * 16 + n;   // 0..31
            v8f c;
            const float bias = bd1[gcol];
#pragma unroll
            for (int r = 0; r < 8; ++r) c[r] = bias;
            v16h b0, b1;
#pragma unroll
            for (int i = 0; i < 16; ++i) {
                int kp = kprime_of(lane, i);
                b0[i] = (_Float16)Wd1[gcol * 64 + kp];
                b1[i] = (_Float16)Wd1[gcol * 64 + 32 + kp];
            }
            c = __builtin_amdgcn_wmma_f32_16x16x32_f16(false, a2, false, b0,
                                                       (short)0, c, false, false);
            c = __builtin_amdgcn_wmma_f32_16x16x32_f16(false, a3, false, b1,
                                                       (short)0, c, false, false);
#pragma unroll
            for (int r = 0; r < 8; ++r) {                  // relu + repack to A layout
                const float yv = fmaxf(c[r], 0.0f);
                const int M  = r + row0;
                const int L  = M + (((gcol >> 3) & 1) << 4);
                const int ii = ((gcol >> 4) << 3) | (gcol & 7);
                sY[L * 16 + ii] = (_Float16)yv;
            }
        }
    }
    __syncthreads();
    if (w == 0) {
        v16h ya = *(const v16h*)&sY[lane * 16];
#pragma unroll
        for (int nt = 0; nt < 2; ++nt) {
            const int oc = nt * 16 + n;                    // valid when oc < 24
            v8f c;
            const float bias = (oc < 24) ? bd2[oc] : 0.0f;
#pragma unroll
            for (int r = 0; r < 8; ++r) c[r] = bias;
            v16h b;
#pragma unroll
            for (int i = 0; i < 16; ++i) {
                int kp = kprime_of(lane, i);
                b[i] = (oc < 24) ? (_Float16)Wd2[oc * 32 + kp] : (_Float16)0.0f;
            }
            c = __builtin_amdgcn_wmma_f32_16x16x32_f16(false, ya, false, b,
                                                       (short)0, c, false, false);
            if (oc < 24) {
#pragma unroll
                for (int r = 0; r < 8; ++r)
                    out[(size_t)(bbase + r + row0) * 24 + oc] = c[r];
            }
        }
    }
}

extern "C" void kernel_launch(void* const* d_in, const int* in_sizes, int n_in,
                              void* d_out, int out_size, void* d_ws, size_t ws_size,
                              hipStream_t stream) {
    const float* x    = (const float*)d_in[0];
    const float* Wih1 = (const float*)d_in[1];
    const float* Whh1 = (const float*)d_in[2];
    const float* bih1 = (const float*)d_in[3];
    const float* bhh1 = (const float*)d_in[4];
    const float* Wih2 = (const float*)d_in[5];
    const float* Whh2 = (const float*)d_in[6];
    const float* bih2 = (const float*)d_in[7];
    const float* bhh2 = (const float*)d_in[8];
    const float* Wd1  = (const float*)d_in[9];
    const float* bd1  = (const float*)d_in[10];
    const float* Wd2  = (const float*)d_in[11];
    const float* bd2  = (const float*)d_in[12];
    float* out = (float*)d_out;

    const int S = 512;
    const int B = in_sizes[0] / S;       // 4096
    dim3 grid(B / 16), block(128);
    hipLaunchKernelGGL(lstm2_fused_kernel, grid, block, 0, stream,
                       x, Wih1, Whh1, bih1, bhh1,
                       Wih2, Whh2, bih2, bhh2,
                       Wd1, bd1, Wd2, bd2, out, S);
}
